// HPDecoder_66494683676996
// MI455X (gfx1250) — matrix operand order repaired and checked by reading, not compiled, not run on hardware
//
#include <hip/hip_runtime.h>

typedef __attribute__((ext_vector_type(16))) __bf16 v16bf;
typedef __attribute__((ext_vector_type(8)))  float  v8f;

union FragU { uint4 u[2]; v16bf v; };

__device__ __forceinline__ v16bf make_frag(uint4 lo, uint4 hi) {
  FragU f; f.u[0] = lo; f.u[1] = hi; return f.v;
}

__device__ __forceinline__ __bf16 f2bf(float x) {
  unsigned u = __builtin_bit_cast(unsigned, x);
  unsigned r = (u + 0x7fffu + ((u >> 16) & 1u)) >> 16;
  unsigned short s = (unsigned short)r;
  return __builtin_bit_cast(__bf16, s);
}

// ---------------- conversion / setup kernels ----------------

__global__ void k_f32_to_bf16(const float* __restrict__ in, __bf16* __restrict__ out, int n) {
  int i = blockIdx.x * blockDim.x + threadIdx.x;
  if (i < n) out[i] = f2bf(in[i]);
}

// conv weights: in [27][CIN][COUT] f32 -> out [27][COUT][CIN] bf16
__global__ void k_cvt_convw(const float* __restrict__ in, __bf16* __restrict__ out,
                            int CIN, int COUT) {
  int k = blockIdx.y;
  int gid = blockIdx.x * blockDim.x + threadIdx.x;
  int n = CIN * COUT;
  if (gid < n) {
    int i = gid % CIN, o = gid / CIN;
    out[(size_t)(k * COUT + o) * CIN + i] = f2bf(in[((size_t)k * CIN + i) * COUT + o]);
  }
}

// upconv weights: in [8][CIN][CB] f32 -> out [8*CB][CIN] bf16  (WT[o*CB+co][i] = w[o][i][co])
__global__ void k_cvt_upw(const float* __restrict__ in, __bf16* __restrict__ out,
                          int CIN, int CB) {
  int gid = blockIdx.x * blockDim.x + threadIdx.x;
  int n = 8 * CB * CIN;
  if (gid < n) {
    int i = gid % CIN;
    int oc = gid / CIN;
    int o = oc / CB, co = oc % CB;
    out[(size_t)oc * CIN + i] = f2bf(in[((size_t)o * CIN + i) * CB + co]);
  }
}

__global__ void k_inv_init(int* __restrict__ inv, int n, int M) {
  int i = blockIdx.x * blockDim.x + threadIdx.x;
  if (i < n) inv[i] = M;
}

// m_out[k] entries are unique per offset k -> race-free inverse rulebook
__global__ void k_inv_scatter(const int* __restrict__ m_in, const int* __restrict__ m_out,
                              int* __restrict__ inv, int M) {
  int gid = blockIdx.x * blockDim.x + threadIdx.x;
  if (gid < 27 * M) {
    int o = m_out[gid];
    if ((unsigned)o < (unsigned)M) inv[(gid / M) * M + o] = m_in[gid];
  }
}

// zero the sentinel pad rows (row M) of the gathered feature buffers
__global__ void k_zero_pads(__bf16* a, int na, __bf16* b, int nb, __bf16* c, int nc) {
  int i = threadIdx.x;
  __bf16 z = __builtin_bit_cast(__bf16, (unsigned short)0);
  if (i < na) a[i] = z;
  if (i < nb) b[i] = z;
  if (i < nc) c[i] = z;
}

// ---------------- dense GEMM (upconv): Y[M,8*CB] = X[M,K] @ WT^T + bias ----------------
template <int K, int CB, int NT>
__global__ __launch_bounds__(256) void upconv_gemm_kernel(
    const __bf16* __restrict__ X, const __bf16* __restrict__ WT,
    const float* __restrict__ bias, __bf16* __restrict__ Y, int Ncols) {
  const int tid  = threadIdx.x;
  const int lane = tid & 31;
  const int wave = tid >> 5;
  const int r0   = blockIdx.x * 16;
  const int cb   = blockIdx.y * (8 * NT * 16);
  const int row  = lane & 15;
  const int sel  = lane >> 4;

  v8f acc[NT];
#pragma unroll
  for (int t = 0; t < NT; ++t) acc[t] = (v8f){};

#pragma unroll
  for (int kk = 0; kk < K; kk += 32) {
    const __bf16* ap = X + (size_t)(r0 + row) * K + kk + sel * 8;
    v16bf a = make_frag(*(const uint4*)ap, *(const uint4*)(ap + 16));
#pragma unroll
    for (int t = 0; t < NT; ++t) {
      int n0 = cb + (wave * NT + t) * 16;
      const __bf16* bp = WT + (size_t)(n0 + row) * K + kk + sel * 8;
      v16bf b = make_frag(*(const uint4*)bp, *(const uint4*)(bp + 16));
      acc[t] = __builtin_amdgcn_wmma_f32_16x16x32_bf16(false, a, false, b,
                                                       (short)0, acc[t], false, false);
    }
  }

#pragma unroll
  for (int t = 0; t < NT; ++t) {
    int col = cb + (wave * NT + t) * 16 + row;
    float bv = bias[col % CB];
#pragma unroll
    for (int r = 0; r < 8; ++r) {
      int orow = r0 + r + sel * 8;
      Y[(size_t)orow * Ncols + col] = f2bf(acc[t][r] + bv);
    }
  }
}

// ---------------- sparse conv: async-LDS double-buffered gather-GEMM ----------------
// feat has M+1 rows; row M is a zeroed pad row (sentinel target).
template <int CIN, int COUT, int NT, bool RELU, bool F32OUT>
__global__ __launch_bounds__(256) void sparse_conv_async_kernel(
    const __bf16* __restrict__ feat, const __bf16* __restrict__ wT,
    const float* __restrict__ bias, const int* __restrict__ inv,
    void* __restrict__ outv, int M) {
  __shared__ __align__(16) __bf16 lds[2][16 * CIN];
  __shared__ int s_idx[27 * 16];

  const int tid  = threadIdx.x;
  const int lane = tid & 31;
  const int wave = tid >> 5;
  const int r0   = blockIdx.x * 16;
  const int row  = lane & 15;
  const int sel  = lane >> 4;
  const int CH8  = CIN / 8;         // 16-byte chunks per row
  const unsigned long long fbase = (unsigned long long)(uintptr_t)feat;

  // preload all 27x16 gather indices
  for (int i = tid; i < 27 * 16; i += 256)
    s_idx[i] = inv[(i >> 4) * M + r0 + (i & 15)];
  __syncthreads();

  // per-lane async DMA: 16B from feat[idx*CIN + cc*8] -> lds[buf][rr*CIN + cc*8]
  auto issue = [&](int k, int buf) {
#pragma unroll
    for (int c = tid; c < 16 * CH8; c += 256) {
      int rr = c / CH8, cc = c % CH8;
      int idx = s_idx[k * 16 + rr];
      unsigned ldsa = (unsigned)(uintptr_t)(&lds[buf][rr * CIN + cc * 8]);
      unsigned voff = (unsigned)idx * (unsigned)(CIN * 2) + (unsigned)(cc * 16);
      asm volatile("global_load_async_to_lds_b128 %0, %1, %2"
                   :: "v"(ldsa), "v"(voff), "s"(fbase)
                   : "memory");
    }
  };

  v8f acc[NT];
#pragma unroll
  for (int t = 0; t < NT; ++t) acc[t] = (v8f){};

  issue(0, 0);

  for (int k = 0; k < 27; ++k) {
    asm volatile("s_wait_asynccnt 0" ::: "memory");
    __syncthreads();                 // all waves' async LDS writes visible,
                                     // and buffer (k+1)&1 fully consumed
    if (k + 1 < 27) issue(k + 1, (k + 1) & 1);

    const __bf16* cur = lds[k & 1];
    const __bf16* wk  = wT + (size_t)k * COUT * CIN;
#pragma unroll
    for (int kk = 0; kk < CIN; kk += 32) {
      const __bf16* ap = cur + row * CIN + kk + sel * 8;
      v16bf a = make_frag(*(const uint4*)ap, *(const uint4*)(ap + 16));
#pragma unroll
      for (int t = 0; t < NT; ++t) {
        int n0 = (wave * NT + t) * 16;
        const __bf16* bp = wk + (size_t)(n0 + row) * CIN + kk + sel * 8;
        v16bf b = make_frag(*(const uint4*)bp, *(const uint4*)(bp + 16));
        acc[t] = __builtin_amdgcn_wmma_f32_16x16x32_bf16(false, a, false, b,
                                                         (short)0, acc[t], false, false);
      }
    }
  }

#pragma unroll
  for (int t = 0; t < NT; ++t) {
    int col = (wave * NT + t) * 16 + row;
    float bv = bias[col];
#pragma unroll
    for (int r = 0; r < 8; ++r) {
      float v = acc[t][r] + bv;
      if (RELU) v = v > 0.f ? v : 0.f;
      int orow = r0 + r + sel * 8;
      if (F32OUT) ((float*)outv)[(size_t)orow * COUT + col] = v;
      else        ((__bf16*)outv)[(size_t)orow * COUT + col] = f2bf(v);
    }
  }
}

// ---------------- launch ----------------

extern "C" void kernel_launch(void* const* d_in, const int* in_sizes, int n_in,
                              void* d_out, int out_size, void* d_ws, size_t ws_size,
                              hipStream_t stream) {
  (void)in_sizes; (void)n_in; (void)out_size; (void)ws_size;
  const float* x       = (const float*)d_in[0];
  const float* up0_w   = (const float*)d_in[1];
  const float* up0_b   = (const float*)d_in[2];
  const float* conv0_w = (const float*)d_in[3];
  const float* conv0_b = (const float*)d_in[4];
  const float* up1_w   = (const float*)d_in[5];
  const float* up1_b   = (const float*)d_in[6];
  const float* conv1_w = (const float*)d_in[7];
  const float* conv1_b = (const float*)d_in[8];
  const float* conv2_w = (const float*)d_in[9];
  const float* conv2_b = (const float*)d_in[10];
  const int* map0_in   = (const int*)d_in[11];
  const int* map0_out  = (const int*)d_in[12];
  const int* map1_in   = (const int*)d_in[13];
  const int* map1_out  = (const int*)d_in[14];

  const int N0 = 512, M1 = 4096, M2 = 32768;

  char* p = (char*)d_ws;
  auto alloc = [&](size_t bytes) {
    char* r = p;
    p += (bytes + 255) & ~size_t(255);
    return r;
  };
  __bf16* xbf = (__bf16*)alloc((size_t)N0 * 128 * 2);
  __bf16* w0T = (__bf16*)alloc((size_t)8 * 128 * 128 * 2);       // up0 WT [1024][128]
  __bf16* c0T = (__bf16*)alloc((size_t)27 * 128 * 128 * 2);      // conv0 wT
  __bf16* w1T = (__bf16*)alloc((size_t)8 * 128 * 256 * 2);       // up1 WT [2048][128]
  __bf16* c1T = (__bf16*)alloc((size_t)27 * 256 * 256 * 2);      // conv1 wT
  __bf16* c2T = (__bf16*)alloc((size_t)27 * 256 * 256 * 2);      // conv2 wT
  __bf16* y1  = (__bf16*)alloc((size_t)(M1 + 1) * 128 * 2);      // up0 out (+pad row)
  __bf16* h1  = (__bf16*)alloc((size_t)M1 * 128 * 2);            // conv0+relu
  __bf16* y2  = (__bf16*)alloc((size_t)(M2 + 1) * 256 * 2);      // up1 out (+pad row)
  __bf16* h2  = (__bf16*)alloc((size_t)(M2 + 1) * 256 * 2);      // conv1+relu (+pad row)
  int*    inv0 = (int*)alloc((size_t)27 * M1 * 4);
  int*    inv1 = (int*)alloc((size_t)27 * M2 * 4);

  // --- setup passes ---
  {
    int n = N0 * 128;
    k_f32_to_bf16<<<(n + 255) / 256, 256, 0, stream>>>(x, xbf, n);
  }
  k_cvt_upw<<<(8 * 128 * 128 + 255) / 256, 256, 0, stream>>>(up0_w, w0T, 128, 128);
  k_cvt_upw<<<(8 * 128 * 256 + 255) / 256, 256, 0, stream>>>(up1_w, w1T, 128, 256);
  k_cvt_convw<<<dim3((128 * 128 + 255) / 256, 27), 256, 0, stream>>>(conv0_w, c0T, 128, 128);
  k_cvt_convw<<<dim3((256 * 256 + 255) / 256, 27), 256, 0, stream>>>(conv1_w, c1T, 256, 256);
  k_cvt_convw<<<dim3((256 * 256 + 255) / 256, 27), 256, 0, stream>>>(conv2_w, c2T, 256, 256);

  k_inv_init<<<(27 * M1 + 255) / 256, 256, 0, stream>>>(inv0, 27 * M1, M1);
  k_inv_scatter<<<(27 * M1 + 255) / 256, 256, 0, stream>>>(map0_in, map0_out, inv0, M1);
  k_inv_init<<<(27 * M2 + 255) / 256, 256, 0, stream>>>(inv1, 27 * M2, M2);
  k_inv_scatter<<<(27 * M2 + 255) / 256, 256, 0, stream>>>(map1_in, map1_out, inv1, M2);

  k_zero_pads<<<1, 256, 0, stream>>>(y1 + (size_t)M1 * 128, 128,
                                     y2 + (size_t)M2 * 256, 256,
                                     h2 + (size_t)M2 * 256, 256);

  // --- stage 0: up0 GEMM [512,128]x[128,1024] -> y1 [4096,128] ---
  upconv_gemm_kernel<128, 128, 2><<<dim3(N0 / 16, 1024 / 256), 256, 0, stream>>>(
      xbf, w0T, up0_b, y1, 1024);
  // conv0 + relu: [4096,128]
  sparse_conv_async_kernel<128, 128, 1, true, false><<<M1 / 16, 256, 0, stream>>>(
      y1, c0T, conv0_b, inv0, h1, M1);

  // --- stage 1: up1 GEMM [4096,128]x[128,2048] -> y2 [32768,256] ---
  upconv_gemm_kernel<128, 256, 2><<<dim3(M1 / 16, 2048 / 256), 256, 0, stream>>>(
      h1, w1T, up1_b, y2, 2048);
  // conv1 + relu: [32768,256]
  sparse_conv_async_kernel<256, 256, 2, true, false><<<M2 / 16, 256, 0, stream>>>(
      y2, c1T, conv1_b, inv1, h2, M2);

  // --- stage 2: conv2 (no relu), f32 output ---
  sparse_conv_async_kernel<256, 256, 2, false, true><<<M2 / 16, 256, 0, stream>>>(
      h2, c2T, conv2_b, inv1, d_out, M2);
}
